// LogicGatedSNN_8778913153516
// MI455X (gfx1250) — matrix operand order repaired and checked by reading, not compiled, not run on hardware
//
#include <hip/hip_runtime.h>
#include <hip/hip_bf16.h>

typedef __attribute__((ext_vector_type(16))) _Float16 v16h;
typedef __attribute__((ext_vector_type(8)))  float    v8f;
typedef __attribute__((ext_vector_type(4)))  float    v4f;

#define IN_F   8192
#define OUT_F  8192
#define KCHUNK 2048          // 4 K-chunks per row-block
#define THRESH 50.0f

// ---------------------------------------------------------------------------
// Phase 0: convert spike_input to f16 once (scratch in d_out trace region)
// ---------------------------------------------------------------------------
__global__ void snn_cvt_x(const float* __restrict__ x, _Float16* __restrict__ xh) {
    int i = blockIdx.x * blockDim.x + threadIdx.x;
    if (i < IN_F) xh[i] = (_Float16)x[i];
}

// ---------------------------------------------------------------------------
// Phase 1: binarized GEMV via v_wmma_f32_16x16x32_f16.
// One wave (32 threads) per (16-row block, 2048-wide K chunk).
// A fragment: 16x32 tile of (syn>50) ? 1 : 0 in f16, per documented A layout:
//   low lanes (0-15):  row = lane,     K = kb+{0..7}  and kb+{16..23}
//   high lanes (16-31):row = lane-16,  K = kb+{8..15} and kb+{24..31}
// B fragment: x chunk broadcast into every N column -> every D column holds
// the same 16 dot-product partials (layout-agnostic in N).
// ---------------------------------------------------------------------------
__global__ __launch_bounds__(32)
void snn_gemv_wmma(const float* __restrict__ W,
                   const _Float16* __restrict__ xh,
                   float* __restrict__ partial) {
    const int wid  = blockIdx.x;       // 0..2047
    const int rb   = wid >> 2;         // row block 0..511
    const int kc   = wid & 3;          // K chunk 0..3
    const int lane = threadIdx.x;      // 0..31
    const int m    = lane & 15;
    const int hi   = lane >> 4;        // 0 or 1

    const float* __restrict__ wrow = W + (long long)(rb * 16 + m) * IN_F;
    const _Float16 one  = (_Float16)1.0f;
    const _Float16 zero = (_Float16)0.0f;

    v8f acc = {};
    const int k0 = kc * KCHUNK;
    for (int kb = k0; kb < k0 + KCHUNK; kb += 32) {
        const float* p0 = wrow + kb + hi * 8;
        v4f w0 = __builtin_nontemporal_load((const v4f*)(p0));       // K+0..3
        v4f w1 = __builtin_nontemporal_load((const v4f*)(p0 + 4));   // K+4..7
        v4f w2 = __builtin_nontemporal_load((const v4f*)(p0 + 16));  // K+16..19
        v4f w3 = __builtin_nontemporal_load((const v4f*)(p0 + 20));  // K+20..23

        v16h a;
        a[0]  = (w0[0] > THRESH) ? one : zero;
        a[1]  = (w0[1] > THRESH) ? one : zero;
        a[2]  = (w0[2] > THRESH) ? one : zero;
        a[3]  = (w0[3] > THRESH) ? one : zero;
        a[4]  = (w1[0] > THRESH) ? one : zero;
        a[5]  = (w1[1] > THRESH) ? one : zero;
        a[6]  = (w1[2] > THRESH) ? one : zero;
        a[7]  = (w1[3] > THRESH) ? one : zero;
        a[8]  = (w2[0] > THRESH) ? one : zero;
        a[9]  = (w2[1] > THRESH) ? one : zero;
        a[10] = (w2[2] > THRESH) ? one : zero;
        a[11] = (w2[3] > THRESH) ? one : zero;
        a[12] = (w3[0] > THRESH) ? one : zero;
        a[13] = (w3[1] > THRESH) ? one : zero;
        a[14] = (w3[2] > THRESH) ? one : zero;
        a[15] = (w3[3] > THRESH) ? one : zero;

        // B: t=0..7 hold K rows (8*hi + t), t=8..15 hold K rows (16 + 8*hi + t-8), all N equal
        typedef __attribute__((ext_vector_type(8))) _Float16 v8h_t;
        v8h_t b0 = *(const v8h_t*)(xh + kb + hi * 8);
        v8h_t b1 = *(const v8h_t*)(xh + kb + 16 + hi * 8);
        v16h b = __builtin_shufflevector(b0, b1, 0,1,2,3,4,5,6,7,8,9,10,11,12,13,14,15);

        acc = __builtin_amdgcn_wmma_f32_16x16x32_f16(
                  false, a, false, b, (short)0, acc, false, false);
    }

    // D layout: lane half hi, VGPR v -> M = v + 8*hi; all N columns identical.
    // Lanes 0 and 16 write the 16 partials for this (row block, K chunk).
    if ((lane & 15) == 0) {
        float* dst = partial + (rb * 64 + kc * 16 + hi * 8);
#pragma unroll
        for (int v = 0; v < 8; ++v) dst[v] = acc[v];
    }
}

// ---------------------------------------------------------------------------
// Phase 2: deterministic partial reduction + LIF neuron update (8192 neurons)
// ---------------------------------------------------------------------------
__global__ void snn_neuron(const float* __restrict__ partial,
                           const float* __restrict__ mp,
                           const float* __restrict__ thr,
                           float* __restrict__ spikes_o,
                           float* __restrict__ vmem_o,
                           float* __restrict__ thr_o) {
    int o = blockIdx.x * blockDim.x + threadIdx.x;
    if (o >= OUT_F) return;
    const float* p = partial + (o >> 4) * 64 + (o & 15);
    float cur = (p[0] + p[16]) + (p[32] + p[48]);
    float t   = thr[o];
    float v   = mp[o] * 0.8f + cur;
    float s   = (v >= t) ? 1.0f : 0.0f;
    spikes_o[o] = s;
    vmem_o[o]   = v * (1.0f - s) * 0.2f;
    float nt = t + (s - 0.05f) * 0.1f;
    thr_o[o] = fminf(fmaxf(nt, 0.5f), 10.0f);
}

// ---------------------------------------------------------------------------
// Phase 3: eligibility-trace streaming update (512 MB of HBM traffic).
// Non-temporal on the 256MB in / 256MB out streams; x stays cache-resident.
// ---------------------------------------------------------------------------
__global__ __launch_bounds__(256)
void snn_trace(const float* __restrict__ trace,
               const float* __restrict__ x,
               const float* __restrict__ spikes,
               float* __restrict__ trace_o) {
    long long idx = (long long)blockIdx.x * blockDim.x + threadIdx.x; // one float4
    int o  = (int)(idx >> 11);        // 2048 float4 per row
    int i4 = (int)(idx & 2047);
    long long e = (long long)o * IN_F + (long long)i4 * 4;

    float s3 = spikes[o] * 3.0f;
    v4f t  = __builtin_nontemporal_load((const v4f*)(trace + e));
    v4f xv = *(const v4f*)(x + i4 * 4);
    v4f r;
#pragma unroll
    for (int j = 0; j < 4; ++j) {
        float v = t[j] * 0.7f + s3 * xv[j];
        r[j] = fminf(fmaxf(v, 0.0f), 10.0f);
    }
    __builtin_nontemporal_store(r, (v4f*)(trace_o + e));
}

// ---------------------------------------------------------------------------
extern "C" void kernel_launch(void* const* d_in, const int* in_sizes, int n_in,
                              void* d_out, int out_size, void* d_ws, size_t ws_size,
                              hipStream_t stream) {
    const float* x   = (const float*)d_in[0];   // [8192]
    const float* syn = (const float*)d_in[1];   // [8192, 8192]
    const float* mp  = (const float*)d_in[2];   // [8192]
    const float* thr = (const float*)d_in[3];   // [8192]
    const float* tr  = (const float*)d_in[4];   // [8192, 8192]

    float* out      = (float*)d_out;
    float* spikes_o = out;                               // [8192]
    float* vmem_o   = out + OUT_F;                       // [8192]
    float* trace_o  = out + 2 * OUT_F;                   // [8192*8192]
    float* thr_o    = trace_o + (long long)OUT_F * IN_F; // [8192]

    // Scratch lives at the head of the trace output region; phase 3 fully
    // overwrites it after phases 1-2 have consumed it. No cross-call state.
    float*    partial = trace_o;                         // 512*4*16 = 32768 f32
    _Float16* xh      = (_Float16*)(trace_o + 32768);    // 8192 f16

    snn_cvt_x   <<<32,    256, 0, stream>>>(x, xh);
    snn_gemv_wmma<<<2048,  32, 0, stream>>>(syn, xh, partial);
    snn_neuron  <<<32,    256, 0, stream>>>(partial, mp, thr, spikes_o, vmem_o, thr_o);
    snn_trace   <<<65536, 256, 0, stream>>>(tr, x, spikes_o, trace_o);
}
